// SCAN_69526930588338
// MI455X (gfx1250) — compile-verified, matrix operand order checked by reading, not run on hardware
//
#include <hip/hip_runtime.h>
#include <hip/hip_bf16.h>

typedef __bf16 bf16;
typedef __attribute__((ext_vector_type(8)))  bf16  v8bf;
typedef __attribute__((ext_vector_type(16))) bf16  v16bf;
typedef __attribute__((ext_vector_type(8)))  float v8f;

#define NB       64      // batch (captions and images)
#define NR       36      // regions per image
#define NRP      48      // regions padded for WMMA
#define IMGD     2048
#define NL       48      // tokens per caption
#define WORDD    300
#define WORDP    320     // padded to multiple of 32 for WMMA K
#define DD       1024
#define LAMBDA_S 9.0f

// ---------------------------------------------------------------------------
// WMMA fragment loader: 16 bf16 per lane = two contiguous 16-byte chunks
// (K offsets [0..7] and [16..23] relative to p; caller adds half*8)
// ---------------------------------------------------------------------------
__device__ __forceinline__ v16bf load_frag(const bf16* __restrict__ p)
{
    v8bf lo = *(const v8bf*)(p);
    v8bf hi = *(const v8bf*)(p + 16);
    v16bf f;
#pragma unroll
    for (int j = 0; j < 8; ++j) { f[j] = lo[j]; f[j + 8] = hi[j]; }
    return f;
}

// ---------------------------------------------------------------------------
// Elementwise prep kernels
// ---------------------------------------------------------------------------
__global__ __launch_bounds__(256)
void zero_bf16_kernel(unsigned int* __restrict__ p, int ndwords)
{
    int idx = blockIdx.x * 256 + threadIdx.x;
    if (idx < ndwords) p[idx] = 0u;
}

__global__ __launch_bounds__(256)
void conv_img_kernel(const float* __restrict__ in, bf16* __restrict__ outp, int n)
{
    int idx = blockIdx.x * 256 + threadIdx.x;
    if (idx < n) outp[idx] = (bf16)in[idx];
}

// W_img (K x N, f32) -> WT (N x K, bf16)
__global__ __launch_bounds__(256)
void conv_wimg_kernel(const float* __restrict__ W, bf16* __restrict__ WT)
{
    int idx = blockIdx.x * 256 + threadIdx.x;
    if (idx >= DD * IMGD) return;
    int nn = idx / IMGD, k = idx % IMGD;
    WT[idx] = (bf16)W[(size_t)k * DD + nn];
}

// captions -> gathered word embeddings, bf16, K padded 300 -> 320 with zeros
__global__ __launch_bounds__(256)
void gather_cap_kernel(const int* __restrict__ captions, const float* __restrict__ emb,
                       bf16* __restrict__ capE)
{
    int idx = blockIdx.x * 256 + threadIdx.x;
    if (idx >= NB * NL * WORDP) return;
    int row = idx / WORDP, k = idx % WORDP;
    float v = 0.0f;
    if (k < WORDD) {
        int token = captions[row];
        v = emb[(size_t)token * WORDD + k];
    }
    capE[idx] = (bf16)v;
}

// W_txt (300 x 1024, f32) -> WtT (1024 x 320, bf16), K padded with zeros
__global__ __launch_bounds__(256)
void conv_wtxt_kernel(const float* __restrict__ W, bf16* __restrict__ WT)
{
    int idx = blockIdx.x * 256 + threadIdx.x;
    if (idx >= DD * WORDP) return;
    int nn = idx / WORDP, k = idx % WORDP;
    WT[idx] = (bf16)((k < WORDD) ? W[(size_t)k * DD + nn] : 0.0f);
}

// ---------------------------------------------------------------------------
// WMMA GEMM + row L2-norm:  out[remap(row)] = l2norm(A(MxK) @ BT(NxK)^T + bias)
// block = 256 threads (8 waves); block computes one 16-row x 1024-col stripe.
// Wave w owns n-tiles w*8 .. w*8+7. Row norms reduced via LDS ds_add_f32.
// group_in/group_out: image rows 36 -> padded 48 remap (cap: 48 -> 48).
// ---------------------------------------------------------------------------
__global__ __launch_bounds__(256)
void gemm_rownorm_kernel(const bf16* __restrict__ A, const bf16* __restrict__ BT,
                         const float* __restrict__ bias, bf16* __restrict__ outp,
                         int K, int group_in, int group_out)
{
    __shared__ float rowss[16];
    int tile = blockIdx.x;
    int t    = threadIdx.x;
    int wave = t >> 5;
    int lane = t & 31;
    int half = lane >> 4;
    int m    = lane & 15;

    if (t < 16) rowss[t] = 0.0f;
    __syncthreads();

    const bf16* arow = A + (size_t)(tile * 16 + m) * K;

    v8f acc[8];
#pragma unroll
    for (int j = 0; j < 8; ++j)
#pragma unroll
        for (int e = 0; e < 8; ++e) acc[j][e] = 0.0f;

    for (int k0 = 0; k0 < K; k0 += 32) {
        __builtin_prefetch(arow + k0 + 256, 0, 0);        // global_prefetch_b8
        v16bf fa = load_frag(arow + k0 + half * 8);
#pragma unroll
        for (int j = 0; j < 8; ++j) {
            int ncol = (wave * 8 + j) * 16 + m;
            v16bf fb = load_frag(BT + (size_t)ncol * K + k0 + half * 8);
            acc[j] = __builtin_amdgcn_wmma_f32_16x16x32_bf16(
                false, fa, false, fb, (short)0, acc[j], false, false);
        }
    }

    // bias add + per-row sum of squares (row-in-tile = v + 8*half, col = ... + n)
    int n = lane & 15;
    float p[8];
#pragma unroll
    for (int v = 0; v < 8; ++v) p[v] = 0.0f;
#pragma unroll
    for (int j = 0; j < 8; ++j) {
        float b = bias[(wave * 8 + j) * 16 + n];
#pragma unroll
        for (int v = 0; v < 8; ++v) {
            acc[j][v] += b;
            p[v] = fmaf(acc[j][v], acc[j][v], p[v]);
        }
    }
#pragma unroll
    for (int v = 0; v < 8; ++v) atomicAdd(&rowss[v + 8 * half], p[v]);
    __syncthreads();

    // scale, convert, scatter with row remap
#pragma unroll
    for (int v = 0; v < 8; ++v) {
        int rtile = v + 8 * half;
        float inv = 1.0f / fmaxf(sqrtf(rowss[rtile]), 1e-12f);
        int grow = tile * 16 + rtile;
        int g = grow / group_in, rl = grow % group_in;
        size_t orow = (size_t)g * group_out + rl;
#pragma unroll
        for (int j = 0; j < 8; ++j)
            outp[orow * DD + (wave * 8 + j) * 16 + n] = (bf16)(acc[j][v] * inv);
    }
}

// ---------------------------------------------------------------------------
// Per-image Gram matrix via WMMA: G = img(48x1024) @ img^T, one wave per image.
// Valid 36x36 block written straight from accumulators.
// ---------------------------------------------------------------------------
__global__ __launch_bounds__(32)
void gram_wmma_kernel(const bf16* __restrict__ imgB, float* __restrict__ G)
{
    int i    = blockIdx.x;
    int lane = threadIdx.x;
    int half = lane >> 4;
    int m    = lane & 15;
    const bf16* base = imgB + (size_t)i * NRP * DD;

    v8f acc[3][3];
#pragma unroll
    for (int mt = 0; mt < 3; ++mt)
#pragma unroll
        for (int nt = 0; nt < 3; ++nt)
#pragma unroll
            for (int e = 0; e < 8; ++e) acc[mt][nt][e] = 0.0f;

    for (int k0 = 0; k0 < DD; k0 += 32) {
        v16bf fr[3];
#pragma unroll
        for (int mt = 0; mt < 3; ++mt)
            fr[mt] = load_frag(base + (size_t)(mt * 16 + m) * DD + k0 + half * 8);
#pragma unroll
        for (int mt = 0; mt < 3; ++mt)
#pragma unroll
            for (int nt = 0; nt < 3; ++nt)
                acc[mt][nt] = __builtin_amdgcn_wmma_f32_16x16x32_bf16(
                    false, fr[mt], false, fr[nt], (short)0, acc[mt][nt],
                    false, false);
    }

    int n = lane & 15;
    float* Gi = G + (size_t)i * NR * NR;
#pragma unroll
    for (int mt = 0; mt < 3; ++mt)
#pragma unroll
        for (int nt = 0; nt < 3; ++nt)
#pragma unroll
            for (int v = 0; v < 8; ++v) {
                int row = mt * 16 + half * 8 + v;
                int col = nt * 16 + n;
                if (row < NR && col < NR) Gi[row * NR + col] = acc[mt][nt][v];
            }
}

// ---------------------------------------------------------------------------
// Fused attention score. One wave32 per (caption c, image i) pair.
// S(48x48) = cap @ img^T via WMMA, softmax, dot = sum attn*S,
// ||ctx||^2 = attn^T G attn, masked mean.
// ---------------------------------------------------------------------------
__global__ __launch_bounds__(32)
void scan_sim_kernel(const bf16* __restrict__ capB, const bf16* __restrict__ imgB,
                     const float* __restrict__ G, const int* __restrict__ cap_lens,
                     float* __restrict__ out)
{
    __shared__ float sS[NRP][NRP + 1];
    __shared__ float sG[NR][NR + 1];
    __shared__ float sA[NL][NR + 1];

    int i = blockIdx.x;
    int c = blockIdx.y;
    int lane = threadIdx.x;

    for (int idx = lane; idx < NR * NR; idx += 32)
        sG[idx / NR][idx % NR] = G[(size_t)i * NR * NR + idx];

    const bf16* capBase = capB + (size_t)c * NL  * DD;
    const bf16* imgBase = imgB + (size_t)i * NRP * DD;

    int half = lane >> 4;
    int m    = lane & 15;

    v8f acc[3][3];
#pragma unroll
    for (int mt = 0; mt < 3; ++mt)
#pragma unroll
        for (int nt = 0; nt < 3; ++nt)
#pragma unroll
            for (int e = 0; e < 8; ++e) acc[mt][nt][e] = 0.0f;

    for (int k0 = 0; k0 < DD; k0 += 32) {
        v16bf fa[3], fb[3];
#pragma unroll
        for (int mt = 0; mt < 3; ++mt)
            fa[mt] = load_frag(capBase + (size_t)(mt * 16 + m) * DD + k0 + half * 8);
#pragma unroll
        for (int nt = 0; nt < 3; ++nt)
            fb[nt] = load_frag(imgBase + (size_t)(nt * 16 + m) * DD + k0 + half * 8);
#pragma unroll
        for (int mt = 0; mt < 3; ++mt)
#pragma unroll
            for (int nt = 0; nt < 3; ++nt)
                acc[mt][nt] = __builtin_amdgcn_wmma_f32_16x16x32_bf16(
                    false, fa[mt], false, fb[nt], (short)0, acc[mt][nt],
                    false, false);
    }

    int n = lane & 15;
#pragma unroll
    for (int mt = 0; mt < 3; ++mt)
#pragma unroll
        for (int nt = 0; nt < 3; ++nt)
#pragma unroll
            for (int v = 0; v < 8; ++v)
                sS[mt * 16 + half * 8 + v][nt * 16 + n] = acc[mt][nt][v];
    __syncthreads();

    float lsum = 0.0f;
    int clen = cap_lens[c];
    for (int l = lane; l < NL; l += 32) {
        float mx = -1e30f;
        for (int r = 0; r < NR; ++r) mx = fmaxf(mx, sS[l][r]);
        float Z = 0.0f;
        for (int r = 0; r < NR; ++r) {
            float e = __expf(LAMBDA_S * (sS[l][r] - mx));
            sA[l][r] = e;
            Z += e;
        }
        float invZ = 1.0f / Z;
        float dot = 0.0f;
        for (int r = 0; r < NR; ++r) {
            float a = sA[l][r] * invZ;
            sA[l][r] = a;
            dot += a * sS[l][r];
        }
        float xn2 = 0.0f;
        for (int r = 0; r < NR; ++r) {
            float tt = 0.0f;
            for (int r2 = 0; r2 < NR; ++r2) tt = fmaf(sA[l][r2], sG[r][r2], tt);
            xn2 = fmaf(sA[l][r], tt, xn2);
        }
        float xn = sqrtf(fmaxf(xn2, 0.0f));
        float rr = dot / fmaxf(xn, 1e-8f);   // cn == 1 after l2norm
        if (l < clen) lsum += rr;
    }
    for (int off = 16; off > 0; off >>= 1)
        lsum += __shfl_down(lsum, off, 32);
    if (lane == 0) out[(size_t)c * NB + i] = lsum / (float)clen;
}

// ---------------------------------------------------------------------------
extern "C" void kernel_launch(void* const* d_in, const int* in_sizes, int n_in,
                              void* d_out, int out_size, void* d_ws, size_t ws_size,
                              hipStream_t stream)
{
    (void)in_sizes; (void)n_in; (void)out_size; (void)ws_size;
    const float* images    = (const float*)d_in[0];
    const int*   captions  = (const int*)  d_in[1];
    const int*   cap_lens  = (const int*)  d_in[2];
    const float* W_img     = (const float*)d_in[3];
    const float* b_img     = (const float*)d_in[4];
    const float* emb_table = (const float*)d_in[5];
    const float* W_txt     = (const float*)d_in[6];
    const float* b_txt     = (const float*)d_in[7];
    float* out = (float*)d_out;

    char* ws = (char*)d_ws;
    size_t off = 0;
    bf16*  imgB  = (bf16*)(ws + off); off += (size_t)NB * NRP * DD * 2;     // 6.29 MB
    bf16*  capB  = (bf16*)(ws + off); off += (size_t)NB * NL  * DD * 2;     // 6.29 MB
    float* G     = (float*)(ws + off); off += (size_t)NB * NR * NR * 4;     // 331 KB
    bf16*  imgA  = (bf16*)(ws + off); off += (size_t)NB * NR * IMGD * 2;    // 9.44 MB
    bf16*  WimgT = (bf16*)(ws + off); off += (size_t)DD * IMGD * 2;         // 4.19 MB
    bf16*  capE  = (bf16*)(ws + off); off += (size_t)NB * NL * WORDP * 2;   // 1.97 MB
    bf16*  WtxtT = (bf16*)(ws + off); off += (size_t)DD * WORDP * 2;        // 0.66 MB

    // --- prep: zero padded imgB, bf16 conversions, transposes, gather ---
    {
        int nd = NB * NRP * DD / 2;                       // bf16 pairs as dwords
        zero_bf16_kernel<<<(nd + 255) / 256, 256, 0, stream>>>((unsigned int*)imgB, nd);
    }
    {
        int n = NB * NR * IMGD;
        conv_img_kernel<<<(n + 255) / 256, 256, 0, stream>>>(images, imgA, n);
    }
    conv_wimg_kernel<<<(DD * IMGD + 255) / 256, 256, 0, stream>>>(W_img, WimgT);
    gather_cap_kernel<<<(NB * NL * WORDP + 255) / 256, 256, 0, stream>>>(captions, emb_table, capE);
    conv_wtxt_kernel<<<(DD * WORDP + 255) / 256, 256, 0, stream>>>(W_txt, WtxtT);

    // --- WMMA projections + row L2-norm ---
    gemm_rownorm_kernel<<<dim3(NB * NR / 16), dim3(256), 0, stream>>>(
        imgA, WimgT, b_img, imgB, IMGD, NR, NRP);         // 144 tiles, K=2048
    gemm_rownorm_kernel<<<dim3(NB * NL / 16), dim3(256), 0, stream>>>(
        capE, WtxtT, b_txt, capB, WORDP, NL, NL);         // 192 tiles, K=320

    // --- Gram matrices (WMMA) ---
    gram_wmma_kernel<<<dim3(NB), dim3(32), 0, stream>>>(imgB, G);

    // --- fused cross-attention scores ---
    scan_sim_kernel<<<dim3(NB, NB), dim3(32), 0, stream>>>(capB, imgB, G, cap_lens, out);
}
